// WindowAttention_71863392796853
// MI455X (gfx1250) — matrix operand order repaired and checked by reading, not compiled, not run on hardware
//
#include <hip/hip_runtime.h>
#include <hip/hip_bf16.h>
#include <cstdint>

// ---------------------------------------------------------------------------
// Swin-style shifted-window attention for MI455X (gfx1250, wave32, WMMA).
//   B=8, N=4096, DIM=384, HEADS=12, HEAD_DIM=32, WS=128, DISP=64
// Pipeline:
//   k_qkv_gemm : rolled-x @ w_qkv  -> qkv (f16, scratch)   [v_wmma_f32_16x16x32_f16]
//   k_attn     : per (b,h,window) QK^T + softmax + PV      [async-to-LDS + WMMA]
//   k_out_gemm : attn_out @ w_out + b_out, roll on store   [async-to-LDS + WMMA]
// LDS tiles are fragment-contiguous: every WMMA operand is 2x ds_load_b128.
// Transposed staging uses 4x4 register blocks -> ds_store_b64 (no b16 scatter).
// ---------------------------------------------------------------------------

typedef __attribute__((ext_vector_type(16))) _Float16 v16h;
typedef __attribute__((ext_vector_type(8)))  _Float16 v8h;
typedef __attribute__((ext_vector_type(4)))  _Float16 v4h;
typedef __attribute__((ext_vector_type(8)))  float    v8f;

#define WMMA_F16(a, b, c) \
  __builtin_amdgcn_wmma_f32_16x16x32_f16(false, (a), false, (b), (short)0, (c), false, false)

// ---------------- fragment loaders (ISA 7.12.2 layouts, wave32) -------------
// A 16x32 f16, tile stored row-major [m][k], ld multiple of 8 (16B rows):
//   lanes 0-15 : M=lane,    halves = K{0..7} ++ K{16..23}
//   lanes 16-31: M=lane-16, halves = K{8..15} ++ K{24..31}
__device__ __forceinline__ v16h load_a_frag(const _Float16* __restrict__ s,
                                            int ld, int lane) {
  const int o = (lane & 15) * ld + (lane >> 4) * 8;
  const v8h lo = *(const v8h*)(s + o);
  const v8h hi = *(const v8h*)(s + o + 16);
  v16h a;
#pragma unroll
  for (int i = 0; i < 8; ++i) { a[i] = lo[i]; a[i + 8] = hi[i]; }
  return a;
}

// B 32x16 f16, tile stored TRANSPOSED [n][k], ld multiple of 8:
//   lanes 0-15 : N=lane,    K{0..7} ++ K{16..23}   lanes 16-31: N=lane-16, K{8..15} ++ K{24..31}
__device__ __forceinline__ v16h load_b_frag(const _Float16* __restrict__ s,
                                            int ld, int lane) {
  const int o = (lane & 15) * ld + (lane >> 4) * 8;
  const v8h lo = *(const v8h*)(s + o);
  const v8h hi = *(const v8h*)(s + o + 16);
  v16h b;
#pragma unroll
  for (int i = 0; i < 8; ++i) { b[i] = lo[i]; b[i + 8] = hi[i]; }
  return b;
}

// ---------------------------------------------------------------------------
// Kernel 1: qkv = roll(x, -64) @ w_qkv       (f32 in, f16 out)
//   C tile = 16(M) x 64(N), 4 waves, K-step 32, K=384, NOUT=1152
// ---------------------------------------------------------------------------
__global__ __launch_bounds__(128)
void k_qkv_gemm(const float* __restrict__ x, const float* __restrict__ wqkv,
                _Float16* __restrict__ qkvh) {
  constexpr int K = 384, NOUT = 1152;
  constexpr int LDA = 40, LDBT = 40;           // halves; 16B-aligned rows
  __shared__ _Float16 As[16 * LDA];            // [m][k]
  __shared__ _Float16 Bt[64 * LDBT];           // [n][k]  (transposed B tile)

  const int t    = threadIdx.x;
  const int lane = t & 31;
  const int wv   = t >> 5;
  const int mBlk = blockIdx.x * 16;
  const int nBlk = blockIdx.y * 64;

  v8f acc = {};

  // A-load: thread owns 4 consecutive k of one row
  const int aRow = (t * 4) >> 5;
  const int aK   = (t * 4) & 31;
  const int mG   = mBlk + aRow;
  const int srcRow = ((mG >> 12) << 12) | (((mG & 4095) + 64) & 4095); // roll -64
  // B-load: thread owns a 4(k) x 4(n) sub-block, transposed in registers
  const int bN = (t & 15) * 4;                 // n0 in [0,64)
  const int bK = (t >> 4) * 4;                 // k0 in [0,32)

  for (int kb = 0; kb < K; kb += 32) {
    {   // stage A: 16x32 f32 -> f16 row-major
      const float4 v = *(const float4*)(x + (size_t)srcRow * 384 + kb + aK);
      As[aRow * LDA + aK + 0] = (_Float16)v.x;
      As[aRow * LDA + aK + 1] = (_Float16)v.y;
      As[aRow * LDA + aK + 2] = (_Float16)v.z;
      As[aRow * LDA + aK + 3] = (_Float16)v.w;
    }
    {   // stage B: 4x4 block load + register transpose + 4x ds_store_b64
      _Float16 hv[4][4];
#pragma unroll
      for (int i = 0; i < 4; ++i) {
        const float4 v = *(const float4*)(wqkv + (size_t)(kb + bK + i) * NOUT + nBlk + bN);
        hv[i][0] = (_Float16)v.x; hv[i][1] = (_Float16)v.y;
        hv[i][2] = (_Float16)v.z; hv[i][3] = (_Float16)v.w;
      }
#pragma unroll
      for (int j = 0; j < 4; ++j) {
        const v4h o = { hv[0][j], hv[1][j], hv[2][j], hv[3][j] };
        *(v4h*)(&Bt[(bN + j) * LDBT + bK]) = o;
      }
    }
    if (kb + 32 < K)   // global_prefetch_b8 of next B tile
      __builtin_prefetch(wqkv + (size_t)(kb + 32 + bK) * NOUT + nBlk + bN, 0, 0);
    __syncthreads();

    const v16h a = load_a_frag(As, LDA, lane);
    const v16h b = load_b_frag(Bt + (wv * 16) * LDBT, LDBT, lane);
    acc = WMMA_F16(a, b, acc);
    __syncthreads();
  }

  // C layout: VGPR r -> M = r + (lane>=16 ? 8 : 0), N = lane&15
  const int hi = lane >> 4, nl = lane & 15;
#pragma unroll
  for (int r = 0; r < 8; ++r) {
    const int m   = mBlk + r + hi * 8;
    const int col = nBlk + wv * 16 + nl;
    qkvh[(size_t)m * NOUT + col] = (_Float16)acc[r];
  }
}

// ---------------------------------------------------------------------------
// Kernel 2: windowed attention, one block per (window p, head h, batch b)
//   Q,K,V tiles [128 x 32] f16; 8 waves each own a 16-row strip.
//   Q/K staged by GLOBAL_LOAD_ASYNC_TO_LDS_B128 (pure f16 copy, ASYNCcnt).
// ---------------------------------------------------------------------------
__global__ __launch_bounds__(256)
void k_attn(const _Float16* __restrict__ qkvh,
            const float* __restrict__ pos_emb,
            _Float16* __restrict__ attnh) {
  constexpr int LDQ = 40, LDK = 40;            // [token][d], 16B-aligned rows
  constexpr int LDVT = 136;                    // V transposed [d][token]
  constexpr int LDP = 136;                     // P [i][j]
  __shared__ _Float16 Qsh[128 * LDQ];          //  10240 B
  __shared__ _Float16 Ksh[128 * LDK];          //  10240 B
  __shared__ _Float16 Vt[32 * LDVT];           //   8704 B
  __shared__ _Float16 Psh[128 * LDP];          //  34816 B
  __shared__ float    possh[256];              //   1024 B   (total 65024 B)

  const int p  = blockIdx.x;   // window  [0,32)
  const int h  = blockIdx.y;   // head    [0,12)
  const int bb = blockIdx.z;   // batch   [0,8)
  const int t    = threadIdx.x;
  const int lane = t & 31;
  const int wv   = t >> 5;

  if (t < 255) possh[t] = pos_emb[t];

  // ---- async-stage Q and K (straight f16 copies): token (w,p) -> n=w*32+p --
  // Each wave: rows [wv*16, wv*16+16). One b128 instr moves 8 rows x 4 parts.
  {
    const int part = lane & 3;                       // 16B chunk within row
#pragma unroll
    for (int half = 0; half < 2; ++half) {
      const int row = wv * 16 + half * 8 + (lane >> 2);
      const uint32_t goff = (uint32_t)((((size_t)(bb * 4096 + row * 32 + p) * 1152)
                                        + h * 32 + part * 8) * 2);
      const uint32_t ldsQ = (uint32_t)(uintptr_t)&Qsh[row * LDQ + part * 8];
      const uint32_t ldsK = (uint32_t)(uintptr_t)&Ksh[row * LDK + part * 8];
      asm volatile("global_load_async_to_lds_b128 %0, %1, %2"
                   :: "v"(ldsQ), "v"(goff), "s"(qkvh) : "memory");
      asm volatile("global_load_async_to_lds_b128 %0, %1, %2 offset:768"
                   :: "v"(ldsK), "v"(goff), "s"(qkvh) : "memory");
    }
  }
  // ---- stage V transposed [d][token]: 4x4 register blocks -> ds_store_b64 --
  {
    const int j0 = (t & 31) * 4;                     // token group
    const int d0 = (t >> 5) * 4;                     // dim group
    _Float16 hv[4][4];
#pragma unroll
    for (int i = 0; i < 4; ++i) {
      const size_t base = (size_t)(bb * 4096 + (j0 + i) * 32 + p) * 1152
                          + (size_t)h * 32 + 768 + d0;
      const v4h v = *(const v4h*)(qkvh + base);
      hv[i][0] = v[0]; hv[i][1] = v[1]; hv[i][2] = v[2]; hv[i][3] = v[3];
    }
#pragma unroll
    for (int jd = 0; jd < 4; ++jd) {
      const v4h o = { hv[0][jd], hv[1][jd], hv[2][jd], hv[3][jd] };
      *(v4h*)(&Vt[(d0 + jd) * LDVT + j0]) = o;
    }
  }
  asm volatile("s_wait_asynccnt 0x0" ::: "memory");  // Q/K async copies done
  __syncthreads();

  // ---- scores: S[16x128] strip; one WMMA per 16-col tile (K-dim = 32) -----
  const v16h qfrag = load_a_frag(Qsh + (wv * 16) * LDQ, LDQ, lane);
  v8f s[8];
#pragma unroll
  for (int jt = 0; jt < 8; ++jt) {
    const v16h kf = load_b_frag(Ksh + (jt * 16) * LDK, LDK, lane);
    v8f z = {};
    s[jt] = WMMA_F16(qfrag, kf, z);
  }

  // ---- scale + rel-pos bias + shift mask ----------------------------------
  constexpr float scale = 0.17677669529663687f;      // 32^-0.5
  const int hi = lane >> 4, nl = lane & 15;
  const float ninf = -__builtin_inff();
#pragma unroll
  for (int jt = 0; jt < 8; ++jt) {
#pragma unroll
    for (int r = 0; r < 8; ++r) {
      const int i = wv * 16 + r + hi * 8;
      const int j = jt * 16 + nl;
      float v = s[jt][r] * scale + possh[j - i + 127];
      if ((i < 64) != (j < 64)) v = ninf;            // shifted-window mask
      s[jt][r] = v;
    }
  }

  // ---- softmax over 128 cols: 8-reg reduce + 16-lane butterfly ------------
#pragma unroll
  for (int r = 0; r < 8; ++r) {
    float mx = s[0][r];
#pragma unroll
    for (int jt = 1; jt < 8; ++jt) mx = fmaxf(mx, s[jt][r]);
#pragma unroll
    for (int off = 1; off < 16; off <<= 1)
      mx = fmaxf(mx, __shfl_xor(mx, off, 32));       // stays within lane half
    float sum = 0.f;
    float e[8];
#pragma unroll
    for (int jt = 0; jt < 8; ++jt) { e[jt] = __expf(s[jt][r] - mx); sum += e[jt]; }
#pragma unroll
    for (int off = 1; off < 16; off <<= 1)
      sum += __shfl_xor(sum, off, 32);
    const float inv = __frcp_rn(sum);
#pragma unroll
    for (int jt = 0; jt < 8; ++jt) s[jt][r] = e[jt] * inv;
  }

  // ---- P -> LDS (f16) [i][j] for the PV matmul ----------------------------
#pragma unroll
  for (int jt = 0; jt < 8; ++jt) {
#pragma unroll
    for (int r = 0; r < 8; ++r) {
      const int i = wv * 16 + r + hi * 8;
      const int j = jt * 16 + nl;
      Psh[i * LDP + j] = (_Float16)s[jt][r];
    }
  }
  __syncthreads();

  // ---- out = P @ V : strip [16 x 32], K-dim 128 = 4 chunks of 32 ----------
  v8f o0 = {}, o1 = {};
#pragma unroll
  for (int c = 0; c < 4; ++c) {
    const v16h pa  = load_a_frag(Psh + (wv * 16) * LDP + c * 32, LDP, lane);
    const v16h vb0 = load_b_frag(Vt + c * 32,                LDVT, lane);
    const v16h vb1 = load_b_frag(Vt + 16 * LDVT + c * 32,    LDVT, lane);
    o0 = WMMA_F16(pa, vb0, o0);
    o1 = WMMA_F16(pa, vb1, o1);
  }

  // ---- store: reference output mapping n' = p*128 + i ---------------------
#pragma unroll
  for (int r = 0; r < 8; ++r) {
    const int i  = wv * 16 + r + hi * 8;
    const int np = p * 128 + i;
    const size_t base = (size_t)(bb * 4096 + np) * 384 + (size_t)h * 32;
    attnh[base + nl]      = (_Float16)o0[r];
    attnh[base + 16 + nl] = (_Float16)o1[r];
  }
}

// ---------------------------------------------------------------------------
// Kernel 3: out = roll(attn @ w_out + b_out, +64)   (f16 A, f32 out)
//   A tile is a pure f16 copy -> GLOBAL_LOAD_ASYNC_TO_LDS_B128.
// ---------------------------------------------------------------------------
__global__ __launch_bounds__(128)
void k_out_gemm(const _Float16* __restrict__ ah, const float* __restrict__ wout,
                const float* __restrict__ bout, float* __restrict__ out) {
  constexpr int K = 384, NOUT = 384;
  constexpr int LDA = 40, LDBT = 40;
  __shared__ _Float16 As[16 * LDA];
  __shared__ _Float16 Bt[64 * LDBT];

  const int t    = threadIdx.x;
  const int lane = t & 31;
  const int wv   = t >> 5;
  const int mBlk = blockIdx.x * 16;
  const int nBlk = blockIdx.y * 64;

  v8f acc = {};

  const int bN = (t & 15) * 4;
  const int bK = (t >> 4) * 4;
  // async A copy: 64 lanes cover 16 rows x 4 parts of 16B
  const int aRow  = t >> 2;
  const int aPart = t & 3;

  for (int kb = 0; kb < K; kb += 32) {
    if (t < 64) {   // stage A via async DMA to LDS (16x32 halves = 1KB)
      const uint32_t ldsA = (uint32_t)(uintptr_t)&As[aRow * LDA + aPart * 8];
      const uint32_t goff =
          (uint32_t)(((size_t)(mBlk + aRow) * 384 + kb + aPart * 8) * 2);
      asm volatile("global_load_async_to_lds_b128 %0, %1, %2"
                   :: "v"(ldsA), "v"(goff), "s"(ah) : "memory");
    }
    {   // stage B f32 -> f16 transposed [n][k], 4x4 register blocks
      _Float16 hv[4][4];
#pragma unroll
      for (int i = 0; i < 4; ++i) {
        const float4 v = *(const float4*)(wout + (size_t)(kb + bK + i) * NOUT + nBlk + bN);
        hv[i][0] = (_Float16)v.x; hv[i][1] = (_Float16)v.y;
        hv[i][2] = (_Float16)v.z; hv[i][3] = (_Float16)v.w;
      }
#pragma unroll
      for (int j = 0; j < 4; ++j) {
        const v4h o = { hv[0][j], hv[1][j], hv[2][j], hv[3][j] };
        *(v4h*)(&Bt[(bN + j) * LDBT + bK]) = o;
      }
    }
    if (kb + 32 < K)
      __builtin_prefetch(wout + (size_t)(kb + 32 + bK) * NOUT + nBlk + bN, 0, 0);
    asm volatile("s_wait_asynccnt 0x0" ::: "memory");
    __syncthreads();

    const v16h a = load_a_frag(As, LDA, lane);
    const v16h b = load_b_frag(Bt + (wv * 16) * LDBT, LDBT, lane);
    acc = WMMA_F16(a, b, acc);
    __syncthreads();
  }

  const int hi = lane >> 4, nl = lane & 15;
  const int col = nBlk + wv * 16 + nl;
  const float bias = bout[col];
#pragma unroll
  for (int r = 0; r < 8; ++r) {
    const int m = mBlk + r + hi * 8;
    const int dst = ((m >> 12) << 12) | (((m & 4095) + 64) & 4095); // roll +64
    __builtin_nontemporal_store(acc[r] + bias, &out[(size_t)dst * 384 + col]);
  }
}

// ---------------------------------------------------------------------------
extern "C" void kernel_launch(void* const* d_in, const int* in_sizes, int n_in,
                              void* d_out, int out_size, void* d_ws, size_t ws_size,
                              hipStream_t stream) {
  (void)in_sizes; (void)n_in; (void)out_size; (void)ws_size;
  const float* x       = (const float*)d_in[0];  // [8,4096,384]
  const float* w_qkv   = (const float*)d_in[1];  // [384,1152]
  const float* pos_emb = (const float*)d_in[2];  // [255]
  const float* w_out   = (const float*)d_in[3];  // [384,384]
  const float* b_out   = (const float*)d_in[4];  // [384]
  float*       out     = (float*)d_out;          // [8,4096,384]

  // scratch partition (f16): qkv 8*4096*1152 (75.5 MB) | attn 8*4096*384 (25.2 MB)
  _Float16* qkvh  = (_Float16*)d_ws;
  _Float16* attnh = (_Float16*)((char*)d_ws + (size_t)8 * 4096 * 1152 * 2);

  // 1) QKV projection with fused roll(-64):  grid = (32768/16, 1152/64)
  k_qkv_gemm<<<dim3(2048, 18), dim3(128), 0, stream>>>(x, w_qkv, qkvh);

  // 2) windowed attention: one block per (window=32, head=12, batch=8)
  k_attn<<<dim3(32, 12, 8), dim3(256), 0, stream>>>(qkvh, pos_emb, attnh);

  // 3) output projection + bias with fused roll(+64): grid = (2048, 384/64)
  k_out_gemm<<<dim3(2048, 6), dim3(128), 0, stream>>>(attnh, w_out, b_out, out);
}
